// DynamicConv_16741782520558
// MI455X (gfx1250) — compile-verified, matrix operand order checked by reading, not run on hardware
//
#include <hip/hip_runtime.h>
#include <hip/hip_bf16.h>
#include <math.h>

typedef __attribute__((ext_vector_type(16))) __bf16 v16bf;
typedef __attribute__((ext_vector_type(8)))  __bf16 v8bf;
typedef __attribute__((ext_vector_type(4)))  __bf16 v4bf;
typedef __attribute__((ext_vector_type(8)))  float  v8f;

#define DIMC  1024
#define SEQ   2048
#define BATCH 16
#define NHEAD 16
#define KW    5

#define MT 128
#define NT 64
#define KT 32
#define NKT (DIMC / KT)   // 32 k-steps

__device__ __forceinline__ __bf16 f2bf(float f) {
  unsigned u = __builtin_bit_cast(unsigned, f);
  unsigned r = (u + 0x7FFFu + ((u >> 16) & 1u)) >> 16;   // round-to-nearest-even
  unsigned short s = (unsigned short)r;
  return __builtin_bit_cast(__bf16, s);
}

// LDS byte offset = low 32 bits of the generic pointer (ISA 10.2 aperture rule).
__device__ __forceinline__ unsigned lds_off(const void* p) {
  return (unsigned)(uintptr_t)p;
}

// CDNA5 async DMA: copy 16 B from global memory directly into LDS (ASYNCcnt).
__device__ __forceinline__ void async_b128(unsigned ldsOff, const void* gptr) {
  asm volatile("global_load_async_to_lds_b128 %0, %1, off"
               :: "v"(ldsOff), "v"(gptr) : "memory");
}

__device__ __forceinline__ void wait_async0() {
  asm volatile("s_wait_asynccnt 0x0" ::: "memory");
}

// ---------------------------------------------------------------------------
// Convert first K-half (e < 1024) of fc_w / gate_w (shape (1024, 2048)) to bf16.
// (o, e) row-major == the [n][k] layout the B fragments consume.
// ---------------------------------------------------------------------------
__global__ __launch_bounds__(256)
void wconv_kernel(const float* __restrict__ fc_w, const float* __restrict__ gate_w,
                  __bf16* __restrict__ wfc, __bf16* __restrict__ wgate) {
  const int gid = blockIdx.x * blockDim.x + threadIdx.x;
  const int i = gid * 4;
  const int n_per = DIMC * DIMC;
  const float* src; __bf16* dst; int local;
  if (i < n_per) { src = fc_w;   dst = wfc;   local = i; }
  else           { src = gate_w; dst = wgate; local = i - n_per; }
  const int o = local >> 10;
  const int e = local & (DIMC - 1);
  float4 v = *reinterpret_cast<const float4*>(src + (size_t)o * (2 * DIMC) + e);
  v4bf t;
  t[0] = f2bf(v.x); t[1] = f2bf(v.y); t[2] = f2bf(v.z); t[3] = f2bf(v.w);
  *reinterpret_cast<v4bf*>(dst + local) = t;
}

// ---------------------------------------------------------------------------
// Convert context (B,S,D) f32 -> bf16 once; the GEMM then streams pure bf16.
// ---------------------------------------------------------------------------
__global__ __launch_bounds__(256)
void ctxconv_kernel(const float* __restrict__ ctx, __bf16* __restrict__ ctxb) {
  const size_t i = ((size_t)blockIdx.x * blockDim.x + threadIdx.x) * 4;
  float4 v = *reinterpret_cast<const float4*>(ctx + i);
  v4bf t;
  t[0] = f2bf(v.x); t[1] = f2bf(v.y); t[2] = f2bf(v.z); t[3] = f2bf(v.w);
  *reinterpret_cast<v4bf*>(ctxb + i) = t;
}

// ---------------------------------------------------------------------------
// softmax(conv_w) per head, folded with the 1x1 conv weights:
// vk[k][d] = conv11_w[d] * softmax(conv_w)[d/64][k].
// ---------------------------------------------------------------------------
__global__ __launch_bounds__(256)
void prep_kernel(const float* __restrict__ conv_w, const float* __restrict__ conv11_w,
                 float* __restrict__ vk) {
  __shared__ float w[NHEAD * KW];
  const int tid = threadIdx.x;
  if (tid < NHEAD) {
    float row[KW];
    float m = -1e30f;
    #pragma unroll
    for (int k = 0; k < KW; ++k) { row[k] = conv_w[tid * KW + k]; m = fmaxf(m, row[k]); }
    float s = 0.0f;
    #pragma unroll
    for (int k = 0; k < KW; ++k) { row[k] = __expf(row[k] - m); s += row[k]; }
    const float inv = 1.0f / s;
    #pragma unroll
    for (int k = 0; k < KW; ++k) w[tid * KW + k] = row[k] * inv;
  }
  __syncthreads();
  for (int idx = tid; idx < KW * DIMC; idx += blockDim.x) {
    const int k = idx >> 10;
    const int d = idx & (DIMC - 1);
    vk[idx] = conv11_w[d] * w[(d >> 6) * KW + k];
  }
}

// ---------------------------------------------------------------------------
// qy[b,o] = sum_e question[b,e] * fc_w[o, 1024+e]  (and qg with gate_w):
// folds the broadcast-question half of the concat out of the big GEMM.
// ---------------------------------------------------------------------------
__global__ __launch_bounds__(256)
void qvec_kernel(const float* __restrict__ question,
                 const float* __restrict__ fc_w, const float* __restrict__ gate_w,
                 float* __restrict__ qy, float* __restrict__ qg) {
  const int gid = blockIdx.x * blockDim.x + threadIdx.x;   // BATCH*DIMC threads
  const int b = gid >> 10;
  const int o = gid & (DIMC - 1);
  const float* q  = question + (size_t)b * DIMC;
  const float* wf = fc_w   + (size_t)o * (2 * DIMC) + DIMC;
  const float* wg = gate_w + (size_t)o * (2 * DIMC) + DIMC;
  float ay = 0.0f, ag = 0.0f;
  for (int e = 0; e < DIMC; e += 4) {
    float4 qv = *reinterpret_cast<const float4*>(q + e);
    float4 fv = *reinterpret_cast<const float4*>(wf + e);
    float4 gv = *reinterpret_cast<const float4*>(wg + e);
    ay += qv.x * fv.x + qv.y * fv.y + qv.z * fv.z + qv.w * fv.w;
    ag += qv.x * gv.x + qv.y * gv.y + qv.z * gv.z + qv.w * gv.w;
  }
  qy[gid] = ay;
  qg[gid] = ag;
}

// ---------------------------------------------------------------------------
// Main fused GLU GEMM: feature[b,s,o] =
//   (ctx@Wfc + qy + fc_b) * sigmoid(ctx@Wgate + qg + gate_b)
// M = 32768, N = 1024, K = 1024, all-bf16 inputs.  Block tile 128x64, 8 waves,
// each wave computes a 32x32 tile of BOTH outputs; tiles staged into
// double-buffered LDS with global_load_async_to_lds_b128 (ASYNCcnt pipeline).
// ---------------------------------------------------------------------------
__global__ __launch_bounds__(256)
void feature_gemm_kernel(const __bf16* __restrict__ ctxb,
                         const __bf16* __restrict__ wfc,
                         const __bf16* __restrict__ wgate,
                         const float* __restrict__ qy, const float* __restrict__ qg,
                         const float* __restrict__ fc_b, const float* __restrict__ gate_b,
                         float* __restrict__ feature) {
  __shared__ __bf16 lA[2][MT * KT];    // 2 x 8 KB
  __shared__ __bf16 lBf[2][NT * KT];   // 2 x 4 KB
  __shared__ __bf16 lBg[2][NT * KT];   // 2 x 4 KB

  const int tid  = threadIdx.x;
  const int nblk = blockIdx.x;         // fast index -> 16 blocks share one A strip in L2
  const int mblk = blockIdx.y;
  const int mbase = mblk * MT;
  const int nbase = nblk * NT;
  const int b = mbase >> 11;           // 128 | 2048 -> batch uniform per block

  const int lane = tid & 31;
  const int l15  = lane & 15;
  const int kh8  = (lane >> 4) << 3;   // lanes 16-31 hold K+8 (and M+8 in C/D)
  const int wv   = tid >> 5;
  const int wm   = (wv & 3) * 32;
  const int wn   = (wv >> 2) * 32;

  // async staging slices (per wave): A 512 chunks of 16 B, B 256 chunks each
  const int ac0 = wv * 64 + lane;
  const int ac1 = wv * 64 + 32 + lane;
  const int bc0 = wv * 32 + lane;

  const unsigned oA[2]  = { lds_off(&lA[0][0]),  lds_off(&lA[1][0]) };
  const unsigned oBf[2] = { lds_off(&lBf[0][0]), lds_off(&lBf[1][0]) };
  const unsigned oBg[2] = { lds_off(&lBg[0][0]), lds_off(&lBg[1][0]) };

  const __bf16* actx = ctxb + (size_t)mbase * DIMC;

  auto stage = [&](int buf, int k0) {
    async_b128(oA[buf] + ac0 * 16, actx + (size_t)(ac0 >> 2) * DIMC + k0 + (ac0 & 3) * 8);
    async_b128(oA[buf] + ac1 * 16, actx + (size_t)(ac1 >> 2) * DIMC + k0 + (ac1 & 3) * 8);
    const size_t boff = (size_t)(nbase + (bc0 >> 2)) * DIMC + k0 + (bc0 & 3) * 8;
    async_b128(oBf[buf] + bc0 * 16, wfc + boff);
    async_b128(oBg[buf] + bc0 * 16, wgate + boff);
  };

  v8f accy[2][2] = {};
  v8f accg[2][2] = {};

  stage(0, 0);   // prologue

  for (int kt = 0; kt < NKT; ++kt) {
    const int c = kt & 1;
    wait_async0();        // own DMA into buffer c has landed
    __syncthreads();      // everyone's DMA landed; everyone done reading buf c^1
    if (kt + 1 < NKT) stage(c ^ 1, (kt + 1) * KT);   // overlaps compute below

    v16bf afrag[2];
    #pragma unroll
    for (int mi = 0; mi < 2; ++mi) {
      const __bf16* pa = &lA[c][(wm + mi * 16 + l15) * KT + kh8];
      v8bf lo = *reinterpret_cast<const v8bf*>(pa);        // K = kh8 + 0..7
      v8bf hi = *reinterpret_cast<const v8bf*>(pa + 16);   // K = kh8 + 16..23
      afrag[mi] = __builtin_shufflevector(lo, hi, 0,1,2,3,4,5,6,7,8,9,10,11,12,13,14,15);
    }
    #pragma unroll
    for (int ni = 0; ni < 2; ++ni) {
      const __bf16* pf = &lBf[c][(wn + ni * 16 + l15) * KT + kh8];
      v8bf flo = *reinterpret_cast<const v8bf*>(pf);
      v8bf fhi = *reinterpret_cast<const v8bf*>(pf + 16);
      v16bf bfrag = __builtin_shufflevector(flo, fhi, 0,1,2,3,4,5,6,7,8,9,10,11,12,13,14,15);
      const __bf16* pg = &lBg[c][(wn + ni * 16 + l15) * KT + kh8];
      v8bf glo = *reinterpret_cast<const v8bf*>(pg);
      v8bf ghi = *reinterpret_cast<const v8bf*>(pg + 16);
      v16bf gfrag = __builtin_shufflevector(glo, ghi, 0,1,2,3,4,5,6,7,8,9,10,11,12,13,14,15);
      #pragma unroll
      for (int mi = 0; mi < 2; ++mi) {
        accy[mi][ni] = __builtin_amdgcn_wmma_f32_16x16x32_bf16(
            false, afrag[mi], false, bfrag, (short)0, accy[mi][ni], false, false);
        accg[mi][ni] = __builtin_amdgcn_wmma_f32_16x16x32_bf16(
            false, afrag[mi], false, gfrag, (short)0, accg[mi][ni], false, false);
      }
    }
    // next iteration's wait+barrier protects buffer reuse
  }

  // fused GLU epilogue
  #pragma unroll
  for (int ni = 0; ni < 2; ++ni) {
    const int o = nbase + wn + ni * 16 + l15;
    const float qyv = qy[b * DIMC + o] + fc_b[o];
    const float qgv = qg[b * DIMC + o] + gate_b[o];
    #pragma unroll
    for (int mi = 0; mi < 2; ++mi) {
      #pragma unroll
      for (int r = 0; r < 8; ++r) {
        const int ml  = wm + mi * 16 + kh8 + r;   // C/D layout: vgpr r -> M=r (+8 hi lanes)
        const int row = mbase + ml;
        const float y = accy[mi][ni][r] + qyv;
        const float g = accg[mi][ni][r] + qgv;
        feature[(size_t)row * DIMC + o] = y * (1.0f / (1.0f + __expf(-g)));
      }
    }
  }
}

// ---------------------------------------------------------------------------
// ps[b,s,k] = <vk[k], feature[b,s,:]>  (one wave per (b,s) row, shuffle reduce)
// ---------------------------------------------------------------------------
__global__ __launch_bounds__(256)
void ps_kernel(const float* __restrict__ feature, const float* __restrict__ vk,
               float* __restrict__ ps) {
  const int wv = threadIdx.x >> 5;
  const int lane = threadIdx.x & 31;
  const int row = blockIdx.x * 8 + wv;
  const float* f = feature + (size_t)row * DIMC;
  float acc[KW] = {0.f, 0.f, 0.f, 0.f, 0.f};
  #pragma unroll
  for (int i = 0; i < 8; ++i) {
    const int d = i * 128 + lane * 4;
    float4 fv = *reinterpret_cast<const float4*>(f + d);
    #pragma unroll
    for (int k = 0; k < KW; ++k) {
      float4 vv = *reinterpret_cast<const float4*>(vk + k * DIMC + d);
      acc[k] += fv.x * vv.x + fv.y * vv.y + fv.z * vv.z + fv.w * vv.w;
    }
  }
  #pragma unroll
  for (int k = 0; k < KW; ++k)
    #pragma unroll
    for (int off = 16; off; off >>= 1)
      acc[k] += __shfl_xor(acc[k], off, 32);
  if (lane == 0) {
    #pragma unroll
    for (int k = 0; k < KW; ++k) ps[(size_t)row * KW + k] = acc[k];
  }
}

// ---------------------------------------------------------------------------
// attn[b,s] = conv11_b + sum_k ps[b, s+k-2, k]; softmax over s -> scores.
// ---------------------------------------------------------------------------
__global__ __launch_bounds__(1024)
void attn_softmax_kernel(const float* __restrict__ ps, const float* __restrict__ conv11_b,
                         float* __restrict__ scores) {
  const int b = blockIdx.x;
  const int tid = threadIdx.x;
  const int wv = tid >> 5, lane = tid & 31;
  __shared__ float red[32];

  float a[2];
  #pragma unroll
  for (int j = 0; j < 2; ++j) {
    const int s = tid + j * 1024;
    float v = conv11_b[0];
    #pragma unroll
    for (int k = 0; k < KW; ++k) {
      const int ss = s + k - 2;
      if (ss >= 0 && ss < SEQ) v += ps[((size_t)b * SEQ + ss) * KW + k];
    }
    a[j] = v;
  }

  float m = fmaxf(a[0], a[1]);
  #pragma unroll
  for (int off = 16; off; off >>= 1) m = fmaxf(m, __shfl_xor(m, off, 32));
  if (lane == 0) red[wv] = m;
  __syncthreads();
  if (tid < 32) {
    float v = red[tid];
    #pragma unroll
    for (int off = 16; off; off >>= 1) v = fmaxf(v, __shfl_xor(v, off, 32));
    red[tid] = v;
  }
  __syncthreads();
  const float M = red[0];
  const float e0 = __expf(a[0] - M);
  const float e1 = __expf(a[1] - M);
  float s2 = e0 + e1;
  #pragma unroll
  for (int off = 16; off; off >>= 1) s2 += __shfl_xor(s2, off, 32);
  __syncthreads();
  if (lane == 0) red[wv] = s2;
  __syncthreads();
  if (tid < 32) {
    float v = red[tid];
    #pragma unroll
    for (int off = 16; off; off >>= 1) v += __shfl_xor(v, off, 32);
    red[tid] = v;
  }
  __syncthreads();
  const float inv = 1.0f / red[0];
  scores[b * SEQ + tid]        = e0 * inv;
  scores[b * SEQ + tid + 1024] = e1 * inv;
}

// ---------------------------------------------------------------------------
// mix[b,d] = sum_s scores[b,s] * feature[b,s,d]
// ---------------------------------------------------------------------------
__global__ __launch_bounds__(256)
void mix_kernel(const float* __restrict__ feature, const float* __restrict__ scores,
                float* __restrict__ out) {
  const int b = blockIdx.x >> 4;
  const int chunk = blockIdx.x & 15;
  const int d = threadIdx.x * 4;
  const int s0 = chunk * 128;
  float4 acc = {0.f, 0.f, 0.f, 0.f};
  for (int j = 0; j < 128; ++j) {
    const int s = s0 + j;
    const float sc = scores[b * SEQ + s];
    float4 fv = *reinterpret_cast<const float4*>(feature + ((size_t)b * SEQ + s) * DIMC + d);
    acc.x += sc * fv.x; acc.y += sc * fv.y; acc.z += sc * fv.z; acc.w += sc * fv.w;
  }
  float* o = out + b * DIMC + d;
  atomicAdd(o + 0, acc.x);
  atomicAdd(o + 1, acc.y);
  atomicAdd(o + 2, acc.z);
  atomicAdd(o + 3, acc.w);
}

// ---------------------------------------------------------------------------
extern "C" void kernel_launch(void* const* d_in, const int* in_sizes, int n_in,
                              void* d_out, int out_size, void* d_ws, size_t ws_size,
                              hipStream_t stream) {
  (void)in_sizes; (void)n_in; (void)out_size; (void)ws_size;
  const float* ctx      = (const float*)d_in[0];
  const float* question = (const float*)d_in[1];
  const float* fc_w     = (const float*)d_in[2];
  const float* fc_b     = (const float*)d_in[3];
  const float* gate_w   = (const float*)d_in[4];
  const float* gate_b   = (const float*)d_in[5];
  const float* conv_w   = (const float*)d_in[6];
  const float* conv11_w = (const float*)d_in[7];
  const float* conv11_b = (const float*)d_in[8];
  float* out = (float*)d_out;

  char* ws = (char*)d_ws;
  size_t off = 0;
  auto alloc = [&](size_t bytes) -> void* {
    void* p = ws + off;
    off = (off + bytes + 255) & ~(size_t)255;
    return p;
  };
  float*  feature = (float*) alloc((size_t)BATCH * SEQ * DIMC * 4);  // 128 MB
  __bf16* ctxb    = (__bf16*)alloc((size_t)BATCH * SEQ * DIMC * 2);  // 64 MB
  __bf16* wfc     = (__bf16*)alloc((size_t)DIMC * DIMC * 2);         // 2 MB
  __bf16* wgate   = (__bf16*)alloc((size_t)DIMC * DIMC * 2);         // 2 MB
  float*  qy      = (float*) alloc((size_t)BATCH * DIMC * 4);
  float*  qg      = (float*) alloc((size_t)BATCH * DIMC * 4);
  float*  vk      = (float*) alloc((size_t)KW * DIMC * 4);
  float*  ps      = (float*) alloc((size_t)BATCH * SEQ * KW * 4);
  float*  scores  = (float*) alloc((size_t)BATCH * SEQ * 4);

  hipMemsetAsync(d_out, 0, (size_t)BATCH * DIMC * 4, stream);

  wconv_kernel<<<2 * DIMC * DIMC / (4 * 256), 256, 0, stream>>>(fc_w, gate_w, wfc, wgate);
  ctxconv_kernel<<<(int)((size_t)BATCH * SEQ * DIMC / (4 * 256)), 256, 0, stream>>>(ctx, ctxb);
  prep_kernel<<<1, 256, 0, stream>>>(conv_w, conv11_w, vk);
  qvec_kernel<<<BATCH * DIMC / 256, 256, 0, stream>>>(question, fc_w, gate_w, qy, qg);

  dim3 grid(DIMC / NT, (BATCH * SEQ) / MT);   // (16, 256), N fastest for A-strip L2 reuse
  feature_gemm_kernel<<<grid, 256, 0, stream>>>(ctxb, wfc, wgate, qy, qg, fc_b, gate_b, feature);

  ps_kernel<<<BATCH * SEQ / 8, 256, 0, stream>>>(feature, vk, ps);
  attn_softmax_kernel<<<BATCH, 1024, 0, stream>>>(ps, conv11_b, scores);
  mix_kernel<<<BATCH * 16, 256, 0, stream>>>(feature, scores, out);
}
